// stackedRNN_52647709114392
// MI455X (gfx1250) — matrix operand order repaired and checked by reading, not compiled, not run on hardware
//
#include <hip/hip_runtime.h>

// ---------------------------------------------------------------------------
// 2-layer LSTM, persistent WMMA kernel for gfx1250 (MI455X).
// SEQ=512, B=64, IN=512, HID=1024, gates=4096.
// bf16 WMMA GEMMs, weights pre-swizzled into WMMA B-operand layout (L2
// resident), activations+weights staged into LDS with async-to-LDS copies,
// triple-buffered, ASYNCcnt-synchronized. Grid-wide barrier per half step.
// ---------------------------------------------------------------------------

typedef __attribute__((ext_vector_type(16))) __bf16 v16bf;
typedef __attribute__((ext_vector_type(8)))  __bf16 v8bf;
typedef __attribute__((ext_vector_type(8)))  float  v8f;

#define SEQ   512
#define BATCH 64
#define IN0   512
#define HID   1024
#define G4    4096
#define NWG   128      // persistent workgroups; each owns 8 hidden units
#define KC0   48       // K chunks of 32 for layer 0 (K = 512+1024)
#define KC1   64       // K chunks of 32 for layer 1 (K = 1024+1024)

#define XBF_N  16777216
#define WP0_N  6291456
#define WP1_N  8388608

// A stage: [buf][kcLocal][row][40 elems] (32 used, rows padded to 80B so the
// 16B A-operand reads hit distinct LDS banks).  B stage: [buf][nb][1024].
typedef unsigned short ABufT[2][64][40];
typedef unsigned short BBufT[2][1024];

#if __has_builtin(__builtin_amdgcn_s_wait_asynccnt)
  #define ASYNC_WAIT(n) __builtin_amdgcn_s_wait_asynccnt(n)
#else
  #define ASYNC_WAIT(n) asm volatile("s_wait_asynccnt " #n ::: "memory")
#endif

// Per-lane async copy of 16 bytes global -> LDS (tracked by ASYNCcnt).
__device__ __forceinline__ void async_g2l_b128(unsigned ldsOff,
                                               const unsigned short* gsrc) {
  asm volatile("global_load_async_to_lds_b128 %0, %1, off"
               :: "v"(ldsOff), "v"(gsrc)
               : "memory");
}

__device__ __forceinline__ unsigned ldsAddr(const void* p) {
  return (unsigned)(uintptr_t)p;   // low 32 bits of generic addr = LDS offset
}

__device__ __forceinline__ unsigned short f2bf(float f) {
  union { float f; unsigned u; } v; v.f = f;
  unsigned r = v.u + 0x7FFFu + ((v.u >> 16) & 1u);   // round-to-nearest-even
  return (unsigned short)(r >> 16);
}

__device__ __forceinline__ float sigmoidf_(float x) {
  return 1.0f / (1.0f + __expf(-x));
}

// Branch-free tanh on the recurrence critical path: saturates to +/-1 when
// __expf(2x) overflows/underflows, avoiding libm's exec-mask branches.
__device__ __forceinline__ float tanhf_(float x) {
  return 1.0f - 2.0f / (__expf(2.0f * x) + 1.0f);
}

// ---------------------------------------------------------------------------
// Prologue kernels
// ---------------------------------------------------------------------------

__global__ void k_f32_to_bf16(const float* __restrict__ src,
                              unsigned short* __restrict__ dst, int n) {
  for (int i = blockIdx.x * blockDim.x + threadIdx.x; i < n;
       i += gridDim.x * blockDim.x)
    dst[i] = f2bf(src[i]);
}

// Pre-swizzle [w_ih | w_hh] (row-major [4096, K]) into per-lane WMMA B layout:
// e = ((((wg*2+nb)*KC + kc)*32 + lane)*16 + j)
//   column n = (v/8)*1024 + wg*8 + (v&7),  v = nb*16 + (lane&15)
//   k        = kc*32 + (j&8)*2 + (lane&16)/2 + (j&7)
__global__ void k_pack_w(const float* __restrict__ wih,
                         const float* __restrict__ whh,
                         unsigned short* __restrict__ dst,
                         int KIN, int KC, int n) {
  for (int e = blockIdx.x * blockDim.x + threadIdx.x; e < n;
       e += gridDim.x * blockDim.x) {
    int j    = e & 15;
    int lane = (e >> 4) & 31;
    int kc   = (e >> 9) % KC;
    int rest = (e >> 9) / KC;
    int nb   = rest & 1;
    int wg   = rest >> 1;
    int v    = nb * 16 + (lane & 15);
    int ncol = ((v >> 3) << 10) + wg * 8 + (v & 7);
    int k    = kc * 32 + ((j & 8) << 1) + ((lane & 16) >> 1) + (j & 7);
    float val = (k < KIN) ? wih[(size_t)ncol * KIN + k]
                          : whh[(size_t)ncol * HID + (k - KIN)];
    dst[e] = f2bf(val);
  }
}

__global__ void k_init(const float* __restrict__ bih0, const float* __restrict__ bhh0,
                       const float* __restrict__ bih1, const float* __restrict__ bhh1,
                       const float* __restrict__ h0,
                       float* __restrict__ bias,
                       unsigned short* __restrict__ hbuf,
                       unsigned* __restrict__ bar) {
  int i = blockIdx.x * blockDim.x + threadIdx.x;   // 131072 threads
  if (i < 2 * G4) {
    int l = i >> 12, n = i & 4095;
    bias[i] = l ? (bih1[n] + bhh1[n]) : (bih0[n] + bhh0[n]);
  }
  if (i < 2 * BATCH * HID) {
    int l = i >> 16, r = i & 65535;
    hbuf[(l * 2 + 0) * 65536 + r] = f2bf(h0[i]);   // read parity for t=0
    hbuf[(l * 2 + 1) * 65536 + r] = 0;
  }
  if (i < 2) bar[i] = 0;
}

// ---------------------------------------------------------------------------
// Device-wide barrier (persistent kernel, 128 resident WGs)
// ---------------------------------------------------------------------------
__device__ __forceinline__ void gridBarrier(unsigned* cnt, unsigned* gen) {
  __syncthreads();
  if (threadIdx.x == 0) {
    __threadfence();
    unsigned g = __hip_atomic_load(gen, __ATOMIC_RELAXED, __HIP_MEMORY_SCOPE_AGENT);
    unsigned a = __hip_atomic_fetch_add(cnt, 1u, __ATOMIC_ACQ_REL, __HIP_MEMORY_SCOPE_AGENT);
    if (a == NWG - 1u) {
      __hip_atomic_store(cnt, 0u, __ATOMIC_RELAXED, __HIP_MEMORY_SCOPE_AGENT);
      __hip_atomic_fetch_add(gen, 1u, __ATOMIC_RELEASE, __HIP_MEMORY_SCOPE_AGENT);
    } else {
      while (__hip_atomic_load(gen, __ATOMIC_ACQUIRE, __HIP_MEMORY_SCOPE_AGENT) == g)
        __builtin_amdgcn_s_sleep(2);
    }
    __threadfence();
  }
  __syncthreads();
}

// ---------------------------------------------------------------------------
// Stage one K-group (64 K values = 2 WMMA chunks) of A and B into LDS buffer
// `buf` via async-to-LDS copies.  A: 8KB coop copy (2x16B per thread).
// B: 4KB coop copy (1x16B per thread), packed layout is copied verbatim.
// ---------------------------------------------------------------------------
__device__ __forceinline__ void stageGroup(
    int g, int buf, int KIN, int KC,
    const unsigned short* __restrict__ xsrc,
    const unsigned short* __restrict__ hold,
    const unsigned short* __restrict__ wslice,
    ABufT* sA, BBufT* sB) {
  const int tid = threadIdx.x;
#pragma unroll
  for (int h = 0; h < 2; ++h) {
    const int c   = tid + h * 256;        // 512 chunks of 16B cover 8KB of A
    const int kcL = c >> 8;
    const int r   = (c >> 2) & 63;
    const int q   = c & 3;
    const int kg  = g * 64 + kcL * 32 + q * 8;
    const unsigned short* src =
        (kg < KIN) ? (xsrc + (size_t)r * KIN + kg)
                   : (hold + (size_t)r * HID + (kg - KIN));
    async_g2l_b128(ldsAddr(&sA[buf][kcL][r][q * 8]), src);
  }
  {
    const int nbb  = tid >> 7;            // 256 chunks of 16B cover 4KB of B
    const int rest = tid & 127;
    const unsigned short* src =
        wslice + ((size_t)nbb * KC + (size_t)g * 2) * 512 + (size_t)rest * 8;
    async_g2l_b128(ldsAddr(&sB[buf][nbb][rest * 8]), src);
  }
}

// ---------------------------------------------------------------------------
// One LSTM layer timestep for this WG's 8 hidden units (32 gate columns).
// 8 waves x (4 Mblk x 2 Nblk) 16x16 tiles, K pipelined 3 groups deep.
// ---------------------------------------------------------------------------
__device__ __forceinline__ void layerStep(
    const int KIN, const int KC,
    const unsigned short* __restrict__ xsrc,   // [64, KIN]  bf16
    const unsigned short* __restrict__ hold,   // [64, 1024] bf16
    const unsigned short* __restrict__ wslice, // packed B operand, this WG
    const float bval, float* creg,
    unsigned short* hnew,                      // [64, 1024] bf16 write slice
    float* __restrict__ yout,                  // layer1 y row, else null
    float* __restrict__ out,
    const int t, const int layer, const int jbase,
    ABufT* sA, BBufT* sB, float* ldsG) {

  const int tid  = threadIdx.x;
  const int lane = tid & 31;
  const int wv   = tid >> 5;
  const int mb   = wv & 3;
  const int nb   = wv >> 2;
  const int col  = lane & 15;
  const int kb   = (lane & 16) ? 8 : 0;
  const int row  = mb * 16 + col;
  const int G    = KC >> 1;                  // 64-K groups

  v8f acc;
#pragma unroll
  for (int r = 0; r < 8; ++r) acc[r] = bval;

  stageGroup(0, 0, KIN, KC, xsrc, hold, wslice, sA, sB);
  stageGroup(1, 1, KIN, KC, xsrc, hold, wslice, sA, sB);

  for (int g = 0; g < G; ++g) {
    if (g + 1 < G) { ASYNC_WAIT(3); } else { ASYNC_WAIT(0); }
    __syncthreads();                          // group g visible; buf (g-1)%3 free
    if (g + 2 < G)
      stageGroup(g + 2, (g + 2) % 3, KIN, KC, xsrc, hold, wslice, sA, sB);
    const int buf = g % 3;
#pragma unroll
    for (int kcL = 0; kcL < 2; ++kcL) {
      v8bf alo = *(const v8bf*)&sA[buf][kcL][row][kb];       // K = kb..kb+7
      v8bf ahi = *(const v8bf*)&sA[buf][kcL][row][kb + 16];  // K = kb+16..+23
      v16bf a  = __builtin_shufflevector(alo, ahi,
                   0, 1, 2, 3, 4, 5, 6, 7, 8, 9, 10, 11, 12, 13, 14, 15);
      v16bf b  = *(const v16bf*)&sB[buf][nb][kcL * 512 + lane * 16];
      acc = __builtin_amdgcn_wmma_f32_16x16x32_bf16(
              false, a, false, b, (short)0, acc, false, false);
    }
  }

  // Scatter 16x16 f32 tile to LDS gates[64][32]: VGPR r -> M = r+8*(lane>=16).
  {
    const int vcol  = nb * 16 + col;
    const int rbase = mb * 16 + ((lane & 16) ? 8 : 0);
#pragma unroll
    for (int r = 0; r < 8; ++r) ldsG[(rbase + r) * 32 + vcol] = acc[r];
  }
  __syncthreads();   // gates ready; also fences sA/sB reuse by next layer

  // Elementwise: 512 (batch, unit) pairs, 2 per thread; c stays in registers.
#pragma unroll
  for (int q = 0; q < 2; ++q) {
    const int p = tid + q * 256;
    const int b = p >> 3, u = p & 7;
    const int j = jbase + u;
    float gi = ldsG[b * 32 + u];
    float gf = ldsG[b * 32 + 8 + u];
    float gg = ldsG[b * 32 + 16 + u];
    float go = ldsG[b * 32 + 24 + u];
    float iv = sigmoidf_(gi);
    float fv = sigmoidf_(gf);
    float gv = tanhf_(gg);
    float ov = sigmoidf_(go);
    float c  = fv * creg[q] + iv * gv;
    float h  = ov * tanhf_(c);
    creg[q]  = c;
    hnew[(size_t)b * HID + j] = f2bf(h);
    if (yout) yout[(size_t)b * HID + j] = h;
    if (t == SEQ - 1) {
      const size_t YN = (size_t)SEQ * BATCH * HID;
      out[YN + (size_t)layer * 65536 + (size_t)b * HID + j] = h;
      out[YN + 131072 + (size_t)layer * 65536 + (size_t)b * HID + j] = c;
    }
  }
}

// ---------------------------------------------------------------------------
// Persistent LSTM kernel: 128 WGs x 256 threads, full 512-step recurrence.
// ---------------------------------------------------------------------------
__global__ __launch_bounds__(256, 1)
void k_lstm(const unsigned short* __restrict__ xbf,
            const unsigned short* __restrict__ wp0,
            const unsigned short* __restrict__ wp1,
            const float* __restrict__ bias,
            unsigned short* __restrict__ hbuf,
            unsigned* __restrict__ bar,
            const float* __restrict__ c0,
            float* __restrict__ out) {
  __shared__ __align__(32) unsigned short sA_[3][2][64][40];  // 30,720 B
  __shared__ __align__(32) unsigned short sB_[3][2][1024];    // 12,288 B
  __shared__ float ldsG[BATCH * 32];                          //  8,192 B
  ABufT* sA = (ABufT*)sA_;
  BBufT* sB = (BBufT*)sB_;

  const int wg    = blockIdx.x;
  const int tid   = threadIdx.x;
  const int lane  = tid & 31;
  const int nb    = (tid >> 5) >> 2;
  const int jbase = wg * 8;

  const int v     = nb * 16 + (lane & 15);
  const int nreal = ((v >> 3) << 10) + jbase + (v & 7);
  const float bv0 = bias[nreal];
  const float bv1 = bias[G4 + nreal];

  const unsigned short* w0 = wp0 + (size_t)wg * (2u * KC0 * 512u);
  const unsigned short* w1 = wp1 + (size_t)wg * (2u * KC1 * 512u);

  float c0r[2], c1r[2];
#pragma unroll
  for (int q = 0; q < 2; ++q) {
    const int p = tid + q * 256;
    const int b = p >> 3, u = p & 7, j = jbase + u;
    c0r[q] = c0[(size_t)b * HID + j];
    c1r[q] = c0[65536 + (size_t)b * HID + j];
  }

  for (int t = 0; t < SEQ; ++t) {
    const int pr = t & 1, pw = pr ^ 1;

    layerStep(IN0, KC0,
              xbf + (size_t)t * (BATCH * IN0),
              hbuf + (0 * 2 + pr) * 65536,
              w0, bv0, c0r,
              hbuf + (0 * 2 + pw) * 65536,
              nullptr, out, t, 0, jbase, sA, sB, ldsG);

    gridBarrier(&bar[0], &bar[1]);   // publish y0_t to all WGPs

    layerStep(HID, KC1,
              hbuf + (0 * 2 + pw) * 65536,
              hbuf + (1 * 2 + pr) * 65536,
              w1, bv1, c1r,
              hbuf + (1 * 2 + pw) * 65536,
              out + (size_t)t * (BATCH * HID),
              out, t, 1, jbase, sA, sB, ldsG);

    gridBarrier(&bar[0], &bar[1]);   // publish h1 before next step
  }
}

// ---------------------------------------------------------------------------
// Host entry
// ---------------------------------------------------------------------------
extern "C" void kernel_launch(void* const* d_in, const int* in_sizes, int n_in,
                              void* d_out, int out_size, void* d_ws, size_t ws_size,
                              hipStream_t stream) {
  (void)in_sizes; (void)n_in; (void)out_size; (void)ws_size;
  const float* x    = (const float*)d_in[0];
  const float* h0   = (const float*)d_in[1];
  const float* c0   = (const float*)d_in[2];
  const float* wih0 = (const float*)d_in[3];
  const float* whh0 = (const float*)d_in[4];
  const float* bih0 = (const float*)d_in[5];
  const float* bhh0 = (const float*)d_in[6];
  const float* wih1 = (const float*)d_in[7];
  const float* whh1 = (const float*)d_in[8];
  const float* bih1 = (const float*)d_in[9];
  const float* bhh1 = (const float*)d_in[10];

  unsigned short* W    = (unsigned short*)d_ws;
  unsigned short* xbf  = W;
  unsigned short* wp0  = W + XBF_N;
  unsigned short* wp1  = wp0 + WP0_N;
  float*          bias = (float*)(wp1 + WP1_N);
  unsigned short* hbuf = (unsigned short*)(bias + 2 * G4);
  unsigned*       bar  = (unsigned*)(hbuf + 4 * 65536);
  float*          out  = (float*)d_out;

  k_f32_to_bf16<<<2048, 256, 0, stream>>>(x, xbf, XBF_N);
  k_pack_w<<<2048, 256, 0, stream>>>(wih0, whh0, wp0, IN0, KC0, WP0_N);
  k_pack_w<<<2048, 256, 0, stream>>>(wih1, whh1, wp1, HID, KC1, WP1_N);
  k_init<<<512, 256, 0, stream>>>(bih0, bhh0, bih1, bhh1, h0, bias, hbuf, bar);
  k_lstm<<<NWG, 256, 0, stream>>>(xbf, wp0, wp1, bias, hbuf, bar, c0, out);
}
